// CoordinateDecoder_22024592294440
// MI455X (gfx1250) — matrix operand order repaired and checked by reading, not compiled, not run on hardware
//
#include <hip/hip_runtime.h>
#include <hip/hip_bf16.h>
#include <math.h>

typedef __attribute__((ext_vector_type(16))) _Float16 v16h;
typedef __attribute__((ext_vector_type(8)))  _Float16 v8h;
typedef __attribute__((ext_vector_type(8)))  float    v8f;

#define NB       8
#define HGT      64
#define WID      64
#define DM       256
#define NPTS     16384
#define INDIM    810
#define K0PAD    832     // 26 * 32
#define MLPW     256
#define ROWS     32      // rows per block (2 M-tiles of 16)
#define XSTRIDE  840     // 832 + 8 pad halfs (1680B rows, 16B aligned)
#define HSTRIDE  264     // 256 + 8 pad halfs (528B rows, 16B aligned)

// ---- workspace layout (bytes) ----
#define WS_L1    0u                                    // 8*32*32*256 f32 = 8388608
#define WS_L2    8388608u                              // 8*16*16*256 f32 = 2097152
#define WS_W0T   10485760u                             // 256*832 f16 = 425984
#define WS_W1T   10911744u                             // 256*256 f16 = 131072
#define WS_W2T   11042816u
#define WS_W3AT  11173888u
#define WS_W3BT  11304960u                             // 256*832 f16 = 425984
// end = 11730944

__device__ __forceinline__ float gelu_tanh(float x) {
    float x3 = x * x * x;
    return 0.5f * x * (1.0f + tanhf(0.7978845608028654f * (x + 0.044715f * x3)));
}

// ---------------- pyramid downsample (box filter) ----------------
__global__ __launch_bounds__(256) void pyramid_kernel(const float* __restrict__ grid,
                                                      float* __restrict__ L1,
                                                      float* __restrict__ L2) {
    int idx = blockIdx.x * 256 + threadIdx.x;
    const int n1 = NB * 32 * 32 * DM;
    const int n2 = NB * 16 * 16 * DM;
    if (idx < n1) {
        int c = idx & 255; int t = idx >> 8;
        int x = t & 31; t >>= 5;
        int y = t & 31; int b = t >> 5;
        const float* base = grid + (((size_t)(b * HGT + 2 * y) * WID) + 2 * x) * DM + c;
        float s = base[0] + base[DM] + base[(size_t)WID * DM] + base[(size_t)WID * DM + DM];
        L1[idx] = 0.25f * s;
    } else if (idx < n1 + n2) {
        int j = idx - n1;
        int c = j & 255; int t = j >> 8;
        int x = t & 15; t >>= 4;
        int y = t & 15; int b = t >> 4;
        float s = 0.f;
        #pragma unroll
        for (int dy = 0; dy < 4; ++dy)
            #pragma unroll
            for (int dx = 0; dx < 4; ++dx)
                s += grid[(((size_t)(b * HGT + 4 * y + dy) * WID) + 4 * x + dx) * DM + c];
        L2[j] = s * 0.0625f;
    }
}

// ---------------- weight transpose + f16 convert ----------------
// w0T[o][k] k<832 ; w1T/w2T/w3aT [o][k] k<256 ; w3bT[o][k] k<832 (skip part)
__global__ __launch_bounds__(256) void weights_kernel(const float* __restrict__ w0,
                                                      const float* __restrict__ w1,
                                                      const float* __restrict__ w2,
                                                      const float* __restrict__ w3,
                                                      _Float16* __restrict__ w0T,
                                                      _Float16* __restrict__ w1T,
                                                      _Float16* __restrict__ w2T,
                                                      _Float16* __restrict__ w3aT,
                                                      _Float16* __restrict__ w3bT) {
    int idx = blockIdx.x * 256 + threadIdx.x;
    const int s0 = MLPW * K0PAD;         // 212992
    const int s1 = MLPW * MLPW;          // 65536
    if (idx < s0) {
        int o = idx / K0PAD, k = idx - o * K0PAD;
        w0T[idx] = (k < INDIM) ? (_Float16)w0[(size_t)k * MLPW + o] : (_Float16)0.0f;
        return;
    }
    idx -= s0;
    if (idx < s1) { int o = idx >> 8, k = idx & 255; w1T[idx] = (_Float16)w1[(size_t)k * MLPW + o]; return; }
    idx -= s1;
    if (idx < s1) { int o = idx >> 8, k = idx & 255; w2T[idx] = (_Float16)w2[(size_t)k * MLPW + o]; return; }
    idx -= s1;
    if (idx < s1) { int o = idx >> 8, k = idx & 255; w3aT[idx] = (_Float16)w3[(size_t)k * MLPW + o]; return; }
    idx -= s1;
    if (idx < s0) {
        int o = idx / K0PAD, k = idx - o * K0PAD;
        w3bT[idx] = (k < INDIM) ? (_Float16)w3[(size_t)(MLPW + k) * MLPW + o] : (_Float16)0.0f;
    }
}

// ---------------- WMMA tile GEMM: 32(M) x 32(N) per wave ----------------
// act: LDS f16 [32][actStride]; wT: global f16 [256][kpad]; wave owns cols [col0, col0+32)
// acc[mt][nt]: mt in {0,1} rows mt*16.., nt in {0,1} cols col0+nt*16..
// Each B fragment load feeds TWO wmma ops (both M-tiles) -> 2x weight reuse.
__device__ __forceinline__ void gemm_tile(const _Float16* act, int actStride, int kChunks,
                                          const _Float16* __restrict__ wT, int kpad,
                                          int col0, int lane, v8f acc[2][2]) {
    const int g = lane >> 4, l15 = lane & 15;
    const _Float16* arow0 = act + l15 * actStride;          // rows 0..15
    const _Float16* arow1 = act + (16 + l15) * actStride;   // rows 16..31
    for (int kc = 0; kc < kChunks; ++kc) {
        const int kbase = kc * 32;
        if (kc + 1 < kChunks)
            __builtin_prefetch(wT + (size_t)(col0 + l15) * kpad + kbase + 32, 0, 1);
        union { v16h v; v8h h[2]; } a0, a1;
        a0.h[0] = *(const v8h*)(arow0 + kbase + 8 * g);
        a0.h[1] = *(const v8h*)(arow0 + kbase + 16 + 8 * g);
        a1.h[0] = *(const v8h*)(arow1 + kbase + 8 * g);
        a1.h[1] = *(const v8h*)(arow1 + kbase + 16 + 8 * g);
        #pragma unroll
        for (int nt = 0; nt < 2; ++nt) {
            const _Float16* bptr = wT + (size_t)(col0 + nt * 16 + l15) * kpad + kbase + 8 * g;
            union { v16h v; uint4 u[2]; } b;
            b.u[0] = *(const uint4*)(bptr);
            b.u[1] = *(const uint4*)(bptr + 16);
            acc[0][nt] = __builtin_amdgcn_wmma_f32_16x16x32_f16(
                false, a0.v, false, b.v, (short)0, acc[0][nt], false, false);
            acc[1][nt] = __builtin_amdgcn_wmma_f32_16x16x32_f16(
                false, a1.v, false, b.v, (short)0, acc[1][nt], false, false);
        }
    }
}

__device__ __forceinline__ void store_act(const v8f acc[2][2], const float* __restrict__ bias,
                                          _Float16* H, int hStride, int col0, int lane) {
    const int g = lane >> 4, l15 = lane & 15;
    #pragma unroll
    for (int nt = 0; nt < 2; ++nt) {
        int col = col0 + nt * 16 + l15;
        float bv = bias[col];
        #pragma unroll
        for (int mt = 0; mt < 2; ++mt) {
            #pragma unroll
            for (int r = 0; r < 8; ++r) {
                int m = mt * 16 + r + 8 * g;  // C layout: VGPR r -> M=r (lanes 0-15), M=8+r (lanes 16-31)
                H[m * hStride + col] = (_Float16)gelu_tanh(acc[mt][nt][r] + bv);
            }
        }
    }
}

// ---------------- fused featurize + MLP ----------------
__global__ __launch_bounds__(256)
void fused_kernel(const float* __restrict__ grid, const float* __restrict__ coords,
                  const float* __restrict__ L1f, const float* __restrict__ L2f,
                  const _Float16* __restrict__ w0T, const _Float16* __restrict__ w1T,
                  const _Float16* __restrict__ w2T, const _Float16* __restrict__ w3aT,
                  const _Float16* __restrict__ w3bT,
                  const float* __restrict__ b0, const float* __restrict__ b1,
                  const float* __restrict__ b2, const float* __restrict__ b3,
                  const float* __restrict__ w_out, const float* __restrict__ b_out,
                  float* __restrict__ out) {
    __shared__ _Float16 Xs[ROWS][XSTRIDE];     // 53760 B
    __shared__ _Float16 H0[ROWS][HSTRIDE];     // 16896 B
    __shared__ _Float16 H1[ROWS][HSTRIDE];     // 16896 B
    __shared__ int   sBase[ROWS * 3][4];       // 1536 B
    __shared__ float sWx[ROWS * 3], sWy[ROWS * 3];

    const int tid = threadIdx.x;
    const int r0 = blockIdx.x * ROWS;          // 32 rows, never straddles a batch (32 | 16384)
    const int bb = r0 >> 14;                   // batch index

    // Phase A: per (row, level) bilinear params
    if (tid < ROWS * 3) {
        int i = tid / 3, L = tid - 3 * i;
        int n = (r0 + i) & (NPTS - 1);
        float cy = coords[2 * n], cx = coords[2 * n + 1];
        int S = 64 >> L;
        float fy = (cy + 1.0f) * 0.5f * (float)(S - 1);
        float fx = (cx + 1.0f) * 0.5f * (float)(S - 1);
        int y0 = (int)floorf(fy), x0 = (int)floorf(fx);
        float wy = fy - (float)y0, wx = fx - (float)x0;
        y0 = y0 < 0 ? 0 : (y0 > S - 1 ? S - 1 : y0);
        x0 = x0 < 0 ? 0 : (x0 > S - 1 ? S - 1 : x0);
        int y1 = y0 + 1 > S - 1 ? S - 1 : y0 + 1;
        int x1 = x0 + 1 > S - 1 ? S - 1 : x0 + 1;
        wy = fminf(fmaxf(wy, 0.f), 1.f);
        wx = fminf(fmaxf(wx, 0.f), 1.f);
        int rb0 = (bb * S + y0) * S, rb1 = (bb * S + y1) * S;
        sBase[tid][0] = (rb0 + x0) * DM;
        sBase[tid][1] = (rb0 + x1) * DM;
        sBase[tid][2] = (rb1 + x0) * DM;
        sBase[tid][3] = (rb1 + x1) * DM;
        sWx[tid] = wx; sWy[tid] = wy;
    }

    // Phase B: positional encoding (42 elems/row) + zero K-padding
    for (int idx = tid; idx < ROWS * 42; idx += 256) {
        int i = idx / 42, c = idx - 42 * i;
        int n = (r0 + i) & (NPTS - 1);
        float val;
        if (c < 2) {
            val = coords[2 * n + c];
        } else {
            int q = c - 2;
            int fi = q >> 2, w = q & 3;
            float cv = coords[2 * n + (w & 1)];
            float f = 3.14159265358979323846f * (float)(1 << fi);
            val = (w >> 1) ? cosf(cv * f) : sinf(cv * f);
        }
        Xs[i][c] = (_Float16)val;
    }
    for (int idx = tid; idx < ROWS * (K0PAD - INDIM); idx += 256) {
        int i = idx / (K0PAD - INDIM), c = INDIM + (idx - i * (K0PAD - INDIM));
        Xs[i][c] = (_Float16)0.0f;
    }
    __syncthreads();

    // Phase C: pyramid sampling -> Xs[i][42 + L*256 + ch]
    for (int idx = tid; idx < ROWS * 3 * DM; idx += 256) {
        int i = idx / (3 * DM), q = idx - i * (3 * DM);
        int L = q >> 8, ch = q & 255;
        int e = i * 3 + L;
        const float* p = (L == 0) ? grid : ((L == 1) ? L1f : L2f);
        float wx = sWx[e], wy = sWy[e];
        float v00 = p[sBase[e][0] + ch], v01 = p[sBase[e][1] + ch];
        float v10 = p[sBase[e][2] + ch], v11 = p[sBase[e][3] + ch];
        float v = (1.f - wy) * ((1.f - wx) * v00 + wx * v01)
                +        wy  * ((1.f - wx) * v10 + wx * v11);
        Xs[i][42 + q] = (_Float16)v;
    }
    __syncthreads();

    const int lane = tid & 31;
    const int wave = tid >> 5;
    const int col0 = wave * 32;
    const v8f vzero = {0.f, 0.f, 0.f, 0.f, 0.f, 0.f, 0.f, 0.f};
    v8f acc[2][2];

    // Layer 0: Xs[32x832] @ w0T -> H0
    acc[0][0] = vzero; acc[0][1] = vzero; acc[1][0] = vzero; acc[1][1] = vzero;
    gemm_tile(&Xs[0][0], XSTRIDE, K0PAD / 32, w0T, K0PAD, col0, lane, acc);
    store_act(acc, b0, &H0[0][0], HSTRIDE, col0, lane);
    __syncthreads();

    // Layer 1: H0 @ w1T -> H1
    acc[0][0] = vzero; acc[0][1] = vzero; acc[1][0] = vzero; acc[1][1] = vzero;
    gemm_tile(&H0[0][0], HSTRIDE, MLPW / 32, w1T, MLPW, col0, lane, acc);
    store_act(acc, b1, &H1[0][0], HSTRIDE, col0, lane);
    __syncthreads();

    // Layer 2: H1 @ w2T -> H0
    acc[0][0] = vzero; acc[0][1] = vzero; acc[1][0] = vzero; acc[1][1] = vzero;
    gemm_tile(&H1[0][0], HSTRIDE, MLPW / 32, w2T, MLPW, col0, lane, acc);
    store_act(acc, b2, &H0[0][0], HSTRIDE, col0, lane);
    __syncthreads();

    // Layer 3: [H0 | Xs] @ w3T -> H1   (two accumulating GEMMs)
    acc[0][0] = vzero; acc[0][1] = vzero; acc[1][0] = vzero; acc[1][1] = vzero;
    gemm_tile(&H0[0][0], HSTRIDE, MLPW / 32, w3aT, MLPW, col0, lane, acc);
    gemm_tile(&Xs[0][0], XSTRIDE, K0PAD / 32, w3bT, K0PAD, col0, lane, acc);
    store_act(acc, b3, &H1[0][0], HSTRIDE, col0, lane);
    __syncthreads();

    // Output head: tanh(H1 @ w_out + b_out), 3 outputs per row
    for (int idx = tid; idx < ROWS * 3; idx += 256) {
        int i = idx / 3, j = idx - 3 * i;
        float s = b_out[j];
        for (int k = 0; k < MLPW; ++k)
            s += (float)H1[i][k] * w_out[k * 3 + j];
        out[(size_t)(r0 + i) * 3 + j] = tanhf(s);
    }
}

extern "C" void kernel_launch(void* const* d_in, const int* in_sizes, int n_in,
                              void* d_out, int out_size, void* d_ws, size_t ws_size,
                              hipStream_t stream) {
    const float* grid   = (const float*)d_in[0];
    const float* coords = (const float*)d_in[1];
    const float* w0 = (const float*)d_in[2];
    const float* b0 = (const float*)d_in[3];
    const float* w1 = (const float*)d_in[4];
    const float* b1 = (const float*)d_in[5];
    const float* w2 = (const float*)d_in[6];
    const float* b2 = (const float*)d_in[7];
    const float* w3 = (const float*)d_in[8];
    const float* b3 = (const float*)d_in[9];
    const float* w_out = (const float*)d_in[10];
    const float* b_out = (const float*)d_in[11];
    float* out = (float*)d_out;

    char* ws = (char*)d_ws;
    float*    L1f  = (float*)(ws + WS_L1);
    float*    L2f  = (float*)(ws + WS_L2);
    _Float16* w0T  = (_Float16*)(ws + WS_W0T);
    _Float16* w1T  = (_Float16*)(ws + WS_W1T);
    _Float16* w2T  = (_Float16*)(ws + WS_W2T);
    _Float16* w3aT = (_Float16*)(ws + WS_W3AT);
    _Float16* w3bT = (_Float16*)(ws + WS_W3BT);

    (void)in_sizes; (void)n_in; (void)out_size; (void)ws_size;

    // 1) build pyramid levels
    {
        int total = NB * 32 * 32 * DM + NB * 16 * 16 * DM;
        pyramid_kernel<<<(total + 255) / 256, 256, 0, stream>>>(grid, L1f, L2f);
    }
    // 2) transpose + convert weights to f16
    {
        int total = MLPW * K0PAD * 2 + MLPW * MLPW * 3;
        weights_kernel<<<(total + 255) / 256, 256, 0, stream>>>(w0, w1, w2, w3,
                                                                w0T, w1T, w2T, w3aT, w3bT);
    }
    // 3) fused featurize + WMMA MLP
    {
        int nblocks = (NB * NPTS) / ROWS;   // 4096
        fused_kernel<<<nblocks, 256, 0, stream>>>(grid, coords, L1f, L2f,
                                                  w0T, w1T, w2T, w3aT, w3bT,
                                                  b0, b1, b2, b3, w_out, b_out, out);
    }
}
